// GAU_46651934769312
// MI455X (gfx1250) — matrix operand order repaired and checked by reading, not compiled, not run on hardware
//
#include <hip/hip_runtime.h>
#include <hip/hip_bf16.h>

// ---------------- problem constants ----------------
#define DIMD   512
#define NSEQ   2048
#define NBATCH 8
#define NROWS  (NBATCH * NSEQ)   // 16384
#define EXPAND 1024
#define SHARED 128
#define PROJ   (2 * EXPAND + SHARED) // 2176
#define HALF   (NSEQ / 2)            // rope half = 1024
#define TSLICE 2079                  // T entries needed per 32-row block

// ---------------- vector types ----------------
typedef __attribute__((ext_vector_type(16))) __bf16 v16bf;
typedef __attribute__((ext_vector_type(8)))  float  v8f;
typedef __attribute__((ext_vector_type(4)))  unsigned int u32x4;

// ---------------- WMMA helper ----------------
__device__ __forceinline__ v8f wmma_bf16(v16bf a, v16bf b, v8f c) {
    // D = A(16x32 bf16) * B(32x16 bf16) + C(16x16 f32)
    return __builtin_amdgcn_wmma_f32_16x16x32_bf16(
        false, a, false, b, (short)0, c, false, false);
}

// ---------------- CDNA5 async LDS copy (ASYNCcnt-tracked DMA) ----------------
__device__ __forceinline__ void async_copy_b128(unsigned lds_addr, const void* gptr) {
    // Each lane copies 16 bytes global -> LDS without touching VGPR data path.
    asm volatile("global_load_async_to_lds_b128 %0, %1, off"
                 :: "v"(lds_addr), "v"((unsigned long long)gptr) : "memory");
}
__device__ __forceinline__ void wait_async0() {
#if __has_builtin(__builtin_amdgcn_s_wait_asynccnt)
    __builtin_amdgcn_s_wait_asynccnt(0);
#else
    asm volatile("s_wait_asynccnt 0x0" ::: "memory");
#endif
}
__device__ __forceinline__ unsigned lds_addr_of(const void* p) {
    // flat LDS address low 32 bits == LDS byte offset (aperture rule)
    return (unsigned)(size_t)p;
}

// A fragment: src row-major [rows x ld], tile rows [row0,row0+16), K [k0,k0+32).
// Lane L holds row row0+(L&15); element i: K = k0 + (L>>4)*8 + (i<8 ? i : 16+i-8)
__device__ __forceinline__ v16bf load_frag_a(const __bf16* src, int ld,
                                             int row0, int k0, int lane) {
    union { v16bf v; u32x4 q[2]; } f;
    const __bf16* p = src + (size_t)(row0 + (lane & 15)) * ld + k0 + ((lane >> 4) * 8);
    f.q[0] = *(const u32x4*)p;
    f.q[1] = *(const u32x4*)(p + 16);
    return f.v;
}

// B fragment: B column j = row (col0+j) of `srcT` (pre-transposed, row-major, ld).
// Lane L holds column L&15; element i: K = k0 + (L>>4)*8 + (i<8 ? i : 16+i-8)
__device__ __forceinline__ v16bf load_frag_b(const __bf16* srcT, int ld,
                                             int col0, int k0, int lane) {
    union { v16bf v; u32x4 q[2]; } f;
    const __bf16* p = srcT + (size_t)(col0 + (lane & 15)) * ld + k0 + ((lane >> 4) * 8);
    f.q[0] = *(const u32x4*)p;
    f.q[1] = *(const u32x4*)(p + 16);
    return f.v;
}

// ---------------- prep kernels ----------------
__global__ __launch_bounds__(256) void prep_w1t(const float* __restrict__ W1,
                                                __bf16* __restrict__ W1T) {
    int idx = blockIdx.x * 256 + threadIdx.x;        // over 512*2176
    if (idx < DIMD * PROJ) {
        int k = idx / PROJ, n = idx % PROJ;
        W1T[(size_t)n * DIMD + k] = (__bf16)W1[idx];
    }
}

__global__ __launch_bounds__(256) void prep_w2t(const float* __restrict__ W2,
                                                __bf16* __restrict__ W2T) {
    int idx = blockIdx.x * 256 + threadIdx.x;        // over 1024*512
    if (idx < EXPAND * DIMD) {
        int k = idx / DIMD, n = idx % DIMD;
        W2T[(size_t)n * EXPAND + k] = (__bf16)W2[idx];
    }
}

// RMSNorm: one block per row of 512
__global__ __launch_bounds__(256) void rmsnorm_kernel(const float* __restrict__ x,
                                                      const float* __restrict__ scale,
                                                      __bf16* __restrict__ xn) {
    __shared__ float red[256];
    int row = blockIdx.x;
    const float* xr = x + (size_t)row * DIMD;
    float s = 0.f;
    for (int i = threadIdx.x; i < DIMD; i += 256) { float v = xr[i]; s += v * v; }
    red[threadIdx.x] = s;
    __syncthreads();
    for (int off = 128; off > 0; off >>= 1) {
        if (threadIdx.x < off) red[threadIdx.x] += red[threadIdx.x + off];
        __syncthreads();
    }
    float k = rsqrtf(red[0] * (1.f / DIMD) + 1e-6f) * scale[0];
    for (int i = threadIdx.x; i < DIMD; i += 256)
        xn[(size_t)row * DIMD + i] = (__bf16)(xr[i] * k);
}

// Toeplitz rope bias: T[d+2047] = sum_j (a1b1+a2b2)cos(d*w_j) + (a1b2-a2b1)sin(d*w_j)
__global__ __launch_bounds__(256) void toeplitz_kernel(const float* __restrict__ ra,
                                                       const float* __restrict__ rb,
                                                       float* __restrict__ T) {
    __shared__ float red[256];
    int dd = blockIdx.x;                          // 0..4094
    float d = (float)(dd - (NSEQ - 1));
    float s = 0.f;
    for (int j = threadIdx.x; j < HALF; j += 256) {
        float a1 = ra[j], a2 = ra[j + HALF];
        float b1 = rb[j], b2 = rb[j + HALF];
        float w = powf(10000.f, -(float)j * (1.f / HALF));
        float sn, cs;
        sincosf(d * w, &sn, &cs);
        s += (a1 * b1 + a2 * b2) * cs + (a1 * b2 - a2 * b1) * sn;
    }
    red[threadIdx.x] = s;
    __syncthreads();
    for (int off = 128; off > 0; off >>= 1) {
        if (threadIdx.x < off) red[threadIdx.x] += red[threadIdx.x + off];
        __syncthreads();
    }
    if (threadIdx.x == 0) T[dd] = red[0];
}

// ---------------- GEMM1: uv = swish(xn @ W1 + b1), fused routing ----------------
// grid (256, 17), block 256 (8 waves). BM=64, BN=128; wave w covers 64x16 cols.
// A tile double-buffered in LDS, filled by async global->LDS DMA.
__global__ __launch_bounds__(256)
void gemm1_kernel(const __bf16* __restrict__ xn, const __bf16* __restrict__ w1t,
                  const float* __restrict__ b1, const float* __restrict__ gamma,
                  const float* __restrict__ beta,
                  __bf16* __restrict__ u, __bf16* __restrict__ vt,
                  __bf16* __restrict__ qb, __bf16* __restrict__ kb) {
    __shared__ __align__(16) __bf16 As[2][64 * 32];
    int w = threadIdx.x >> 5, lane = threadIdx.x & 31;
    int row0 = blockIdx.x * 64;
    int col0 = blockIdx.y * 128 + w * 16;
    int t = threadIdx.x, rr = t >> 2, cc = (t & 3) * 8;
    const __bf16* gsrc = xn + (size_t)(row0 + rr) * DIMD + cc;
    unsigned lds0 = lds_addr_of(&As[0][rr * 32 + cc]);
    unsigned lds1 = lds_addr_of(&As[1][rr * 32 + cc]);

    async_copy_b128(lds0, gsrc);                    // prefetch tile 0
    v8f acc[4] = {};
    int buf = 0;
    for (int k0 = 0; k0 < DIMD; k0 += 32) {
        wait_async0();
        __syncthreads();                            // tile `buf` ready, other buffer free
        if (k0 + 32 < DIMD)
            async_copy_b128(buf ? lds0 : lds1, gsrc + k0 + 32);
        v16bf bf = load_frag_b(w1t, DIMD, col0, k0, lane);
        for (int rt = 0; rt < 4; ++rt) {
            v16bf af = load_frag_a(As[buf], 32, rt * 16, 0, lane);
            acc[rt] = wmma_bf16(af, bf, acc[rt]);
        }
        buf ^= 1;
    }
    int col = col0 + (lane & 15);
    float bias = b1[col];
    for (int rt = 0; rt < 4; ++rt) {
        for (int e = 0; e < 8; ++e) {
            int row = row0 + rt * 16 + e + ((lane >> 4) * 8);
            float xv = acc[rt][e] + bias;
            float sw = xv / (1.f + __expf(-xv));         // swish
            if (col < EXPAND) {
                u[(size_t)row * EXPAND + col] = (__bf16)sw;
            } else if (col < 2 * EXPAND) {
                int b = row >> 11, seq = row & (NSEQ - 1);
                vt[((size_t)(b * EXPAND + (col - EXPAND))) * NSEQ + seq] = (__bf16)sw;
            } else {
                int c = col - 2 * EXPAND;                 // 0..127
                qb[(size_t)row * SHARED + c] = (__bf16)(sw * gamma[c] + beta[c]);
                kb[(size_t)row * SHARED + c] = (__bf16)(sw * gamma[SHARED + c] + beta[SHARED + c]);
            }
        }
    }
}

// ---------------- attention: o = u * (relu(q@k^T/N + T)^2 @ v) ----------------
// grid 512, block 256 (8 waves). 32-row tile, loop over 64-col k/v tiles.
// Wave w: S subtile (r=w>>2, c=w&3); then E-slice [w*128, w*128+128).
__global__ __launch_bounds__(256)
void attn_kernel(const __bf16* __restrict__ qb, const __bf16* __restrict__ kb,
                 const __bf16* __restrict__ vt, const __bf16* __restrict__ u,
                 const float* __restrict__ T, __bf16* __restrict__ o) {
    __shared__ __align__(16) __bf16 P[32 * 64];
    __shared__ float Tl[TSLICE + 1];
    int w = threadIdx.x >> 5, lane = threadIdx.x & 31;
    int row0 = blockIdx.x * 32;
    int b = row0 >> 11;
    int m0 = row0 & (NSEQ - 1);
    int r = w >> 2, c = w & 3;

    // stage the T slice this block can touch: d = m0 + (mi - npos + 2047), idx in [0,2078]
    for (int i = threadIdx.x; i < TSLICE; i += 256) Tl[i] = T[m0 + i];

    v16bf qf[4];
    for (int ks = 0; ks < 4; ++ks)
        qf[ks] = load_frag_a(qb, SHARED, row0 + r * 16, ks * 32, lane);

    v8f acc[2][8] = {};
    const __bf16* vbase = vt + (size_t)b * EXPAND * NSEQ;
    const __bf16* kbase = kb + (size_t)b * NSEQ * SHARED;
    __syncthreads();                                   // Tl visible

    for (int n0 = 0; n0 < NSEQ; n0 += 64) {
        // S subtile = q(16x128) @ k^T(128x16)
        v8f s = {};
        for (int ks = 0; ks < 4; ++ks) {
            v16bf kf = load_frag_b(kbase, SHARED, n0 + c * 16, ks * 32, lane);
            s = wmma_bf16(qf[ks], kf, s);
        }
        // bias + relu^2 -> LDS (bf16)
        int ni = c * 16 + (lane & 15);
        int npos = n0 + ni;
        for (int e = 0; e < 8; ++e) {
            int mi = r * 16 + e + ((lane >> 4) * 8);
            float val = s[e] * (1.f / NSEQ) + Tl[mi - npos + (NSEQ - 1)];
            val = fmaxf(val, 0.f);
            val *= val;
            P[mi * 64 + ni] = (__bf16)val;
        }
        __syncthreads();
        // O_slice += P(32x64) @ V(64x128)
        for (int ks = 0; ks < 2; ++ks) {
            v16bf pa0 = load_frag_a(P, 64, 0,  ks * 32, lane);
            v16bf pa1 = load_frag_a(P, 64, 16, ks * 32, lane);
            for (int ct = 0; ct < 8; ++ct) {
                v16bf vb = load_frag_b(vbase, NSEQ, w * 128 + ct * 16, n0 + ks * 32, lane);
                acc[0][ct] = wmma_bf16(pa0, vb, acc[0][ct]);
                acc[1][ct] = wmma_bf16(pa1, vb, acc[1][ct]);
            }
        }
        __syncthreads();
    }
    // epilogue: o = u * acc
    for (int rt = 0; rt < 2; ++rt)
        for (int ct = 0; ct < 8; ++ct)
            for (int e = 0; e < 8; ++e) {
                int row = row0 + rt * 16 + e + ((lane >> 4) * 8);
                int colE = w * 128 + ct * 16 + (lane & 15);
                float uval = (float)u[(size_t)row * EXPAND + colE];
                o[(size_t)row * EXPAND + colE] = (__bf16)(acc[rt][ct][e] * uval);
            }
}

// ---------------- GEMM2: out = o @ W2 + b2 + shortcut ----------------
// grid (256, 4), block 256. BM=64, BN=128. Async double-buffered A tile.
__global__ __launch_bounds__(256)
void gemm2_kernel(const __bf16* __restrict__ oin, const __bf16* __restrict__ w2t,
                  const float* __restrict__ b2, const float* __restrict__ x,
                  float* __restrict__ out) {
    __shared__ __align__(16) __bf16 As[2][64 * 32];
    int w = threadIdx.x >> 5, lane = threadIdx.x & 31;
    int row0 = blockIdx.x * 64;
    int col0 = blockIdx.y * 128 + w * 16;
    int t = threadIdx.x, rr = t >> 2, cc = (t & 3) * 8;
    const __bf16* gsrc = oin + (size_t)(row0 + rr) * EXPAND + cc;
    unsigned lds0 = lds_addr_of(&As[0][rr * 32 + cc]);
    unsigned lds1 = lds_addr_of(&As[1][rr * 32 + cc]);

    async_copy_b128(lds0, gsrc);
    v8f acc[4] = {};
    int buf = 0;
    for (int k0 = 0; k0 < EXPAND; k0 += 32) {
        wait_async0();
        __syncthreads();
        if (k0 + 32 < EXPAND)
            async_copy_b128(buf ? lds0 : lds1, gsrc + k0 + 32);
        v16bf bf = load_frag_b(w2t, EXPAND, col0, k0, lane);
        for (int rt = 0; rt < 4; ++rt) {
            v16bf af = load_frag_a(As[buf], 32, rt * 16, 0, lane);
            acc[rt] = wmma_bf16(af, bf, acc[rt]);
        }
        buf ^= 1;
    }
    int col = col0 + (lane & 15);
    float bias = b2[col];
    for (int rt = 0; rt < 4; ++rt)
        for (int e = 0; e < 8; ++e) {
            int row = row0 + rt * 16 + e + ((lane >> 4) * 8);
            size_t idx = (size_t)row * DIMD + col;
            out[idx] = acc[rt][e] + bias + x[idx];
        }
}

// ---------------- launcher ----------------
extern "C" void kernel_launch(void* const* d_in, const int* in_sizes, int n_in,
                              void* d_out, int out_size, void* d_ws, size_t ws_size,
                              hipStream_t stream) {
    const float* x     = (const float*)d_in[0];
    const float* W1    = (const float*)d_in[1];
    const float* b1    = (const float*)d_in[2];
    const float* W2    = (const float*)d_in[3];
    const float* b2    = (const float*)d_in[4];
    const float* ra    = (const float*)d_in[5];
    const float* rb    = (const float*)d_in[6];
    const float* gamma = (const float*)d_in[7];
    const float* beta  = (const float*)d_in[8];
    const float* nsc   = (const float*)d_in[9];
    float* out = (float*)d_out;

    char* ws = (char*)d_ws;
    size_t off = 0;
    auto alloc = [&](size_t bytes) {
        void* p = ws + off;
        off = (off + bytes + 255) & ~(size_t)255;
        return p;
    };
    __bf16* xn  = (__bf16*)alloc((size_t)NROWS * DIMD * 2);
    __bf16* w1t = (__bf16*)alloc((size_t)PROJ * DIMD * 2);
    __bf16* w2t = (__bf16*)alloc((size_t)DIMD * EXPAND * 2);
    __bf16* u   = (__bf16*)alloc((size_t)NROWS * EXPAND * 2);
    __bf16* vt  = (__bf16*)alloc((size_t)NBATCH * EXPAND * NSEQ * 2);
    __bf16* qb  = (__bf16*)alloc((size_t)NROWS * SHARED * 2);
    __bf16* kb  = (__bf16*)alloc((size_t)NROWS * SHARED * 2);
    float*  T   = (float*) alloc((size_t)(2 * NSEQ - 1) * 4);
    __bf16* o   = (__bf16*)alloc((size_t)NROWS * EXPAND * 2);

    prep_w1t<<<(DIMD * PROJ + 255) / 256, 256, 0, stream>>>(W1, w1t);
    prep_w2t<<<(EXPAND * DIMD + 255) / 256, 256, 0, stream>>>(W2, w2t);
    rmsnorm_kernel<<<NROWS, 256, 0, stream>>>(x, nsc, xn);
    toeplitz_kernel<<<2 * NSEQ - 1, 256, 0, stream>>>(ra, rb, T);

    gemm1_kernel<<<dim3(NROWS / 64, PROJ / 128), 256, 0, stream>>>(
        xn, w1t, b1, gamma, beta, u, vt, qb, kb);

    attn_kernel<<<NROWS / 32, 256, 0, stream>>>(qb, kb, vt, u, T, o);

    gemm2_kernel<<<dim3(NROWS / 64, DIMD / 128), 256, 0, stream>>>(
        o, w2t, b2, x, out);
}
